// WGAAE_Encoder_75935021793690
// MI455X (gfx1250) — matrix operand order repaired and checked by reading, compiled, not run on hardware
//
#include <hip/hip_runtime.h>
#include <math.h>

typedef __bf16 bf16_t;
typedef bf16_t v16bf __attribute__((ext_vector_type(16)));
typedef float  v8f   __attribute__((ext_vector_type(8)));

#define HEADS      4
#define SLOPE      0.2f
#define REAL_MINF  2.2e-10f
#define THETA_MAXF 1000.0f
#define KMINF      0.1f
#define KMAXF      1000.0f
#define NTILES     4          // 16x64 output tile per wave

// ---------- helpers ----------
__device__ __forceinline__ float softplusf(float x) {
  return x > 20.f ? x : __logf(1.f + __expf(x));
}

// monotonic float <-> uint mapping for atomicMax on floats
__device__ __forceinline__ unsigned fmap(float f) {
  unsigned u = __float_as_uint(f);
  return (u >> 31) ? ~u : (u | 0x80000000u);
}
__device__ __forceinline__ float funmap(unsigned u) {
  return __uint_as_float((u >> 31) ? (u & 0x7FFFFFFFu) : ~u);
}

__device__ __forceinline__ float4 ld4(const float* p) {
  return *reinterpret_cast<const float4*>(p);
}

// ---------------------------------------------------------------------------
// f32 -> bf16 streaming convert (8 elements / thread, b128-in / b128-out)
// ---------------------------------------------------------------------------
__global__ void k_f2bf(const float* __restrict__ in, bf16_t* __restrict__ out, int n)
{
  int i = (blockIdx.x * blockDim.x + threadIdx.x) * 8;
  if (i + 7 >= n) return;               // all sizes are multiples of 8
  float4 a = ld4(in + i), b = ld4(in + i + 4);
  union { bf16_t h[8]; uint4 q; } u;
  u.h[0]=(bf16_t)a.x; u.h[1]=(bf16_t)a.y; u.h[2]=(bf16_t)a.z; u.h[3]=(bf16_t)a.w;
  u.h[4]=(bf16_t)b.x; u.h[5]=(bf16_t)b.y; u.h[6]=(bf16_t)b.z; u.h[7]=(bf16_t)b.w;
  *reinterpret_cast<uint4*>(out + i) = u.q;
}

// ---------------------------------------------------------------------------
// GEMM: out[M,On] = act(A[M,K] @ W[On,K]^T + bias)
// A, W pre-converted to bf16. block = 128 threads (4 waves); each wave owns
// a 16x64 output tile (one A fragment reused across 4 WMMAs). K fully
// unrolled; fragments loaded as raw b128 pairs -> zero VALU in the hot loop.
// ---------------------------------------------------------------------------
template<int K>
__global__ void k_gemm_bf16wmma(const bf16_t* __restrict__ A,
                                const bf16_t* __restrict__ W,
                                const float* __restrict__ bias,
                                float* __restrict__ out,
                                int M, int On, int act)
{
  const int lane   = threadIdx.x & 31;
  const int wave   = threadIdx.x >> 5;
  const int mtile  = blockIdx.x * 4 + wave;
  const int ntile0 = blockIdx.y * NTILES;
  if (mtile * 16 >= M) return;          // wave-uniform: EXEC stays all-ones

  const int half = lane >> 4;           // 0: lanes 0-15, 1: lanes 16-31
  const int l15  = lane & 15;
  const int rowA = mtile * 16 + l15;    // A row for this lane
  const int oA   = half ? 8  : 0;       // A K-offset per ISA layout
  const int oB   = half ? 8  : 0;       // B K-offset per ISA layout
  const bf16_t* arow = A + (size_t)rowA * K;

  // per-lane B row base pointers (wave-invariant stride)
  const bf16_t* brow[NTILES];
  #pragma unroll
  for (int t = 0; t < NTILES; ++t)
    brow[t] = W + (size_t)((ntile0 + t) * 16 + l15) * K + oB;

  v8f acc[NTILES];
  #pragma unroll
  for (int t = 0; t < NTILES; ++t) { v8f z = {}; acc[t] = z; }

  union F { uint4 q[2]; v16bf v; };

  #pragma unroll
  for (int kb = 0; kb < K; kb += 32) {
    // ---- load phase: 2 b128 for A, 2 b128 per B tile (all independent) ----
    const bf16_t* ar = arow + kb + oA;
    F af;
    af.q[0] = *reinterpret_cast<const uint4*>(ar);        // elems 0-7  (K+0..7)
    af.q[1] = *reinterpret_cast<const uint4*>(ar + 16);   // elems 8-15 (K+16..23)
    F bf[NTILES];
    #pragma unroll
    for (int t = 0; t < NTILES; ++t) {
      const bf16_t* br = brow[t] + kb;                    // elems 0-7 (K+0..7), 8-15 (K+16..23)
      bf[t].q[0] = *reinterpret_cast<const uint4*>(br);
      bf[t].q[1] = *reinterpret_cast<const uint4*>(br + 16);
    }
    // ---- mma phase ----
    #pragma unroll
    for (int t = 0; t < NTILES; ++t) {
      acc[t] = __builtin_amdgcn_wmma_f32_16x16x32_bf16(
          /*neg_a=*/false, af.v, /*neg_b=*/false, bf[t].v,
          /*c_mod=*/(short)0, acc[t], /*reuse_a=*/false, /*reuse_b=*/false);
    }
  }

  const int mbase = mtile * 16 + half * 8;      // C/D layout: rows r / 8+r
  #pragma unroll
  for (int t = 0; t < NTILES; ++t) {
    const int colB = (ntile0 + t) * 16 + l15;
    const float bb = bias ? bias[colB] : 0.f;
    #pragma unroll
    for (int r = 0; r < 8; ++r) {
      float v = acc[t][r] + bb;
      if (act) v = softplusf(v);
      out[(size_t)(mbase + r) * On + colB] = v;
    }
  }
}

// ---------------------------------------------------------------------------
// GAT helpers
// ---------------------------------------------------------------------------
__global__ void k_srcdst(const float* __restrict__ xw,
                         const float* __restrict__ asrc,
                         const float* __restrict__ adst,
                         float* __restrict__ s, float* __restrict__ d,
                         int N, int On)
{
  int idx = blockIdx.x * blockDim.x + threadIdx.x;
  if (idx >= N * HEADS) return;
  int n = idx >> 2, h = idx & 3;
  int C = On >> 2;
  const float* xr = xw + (size_t)n * On + h * C;
  const float* as = asrc + h * C;
  const float* ad = adst + h * C;
  float ss = 0.f, dd = 0.f;
  for (int c = 0; c < C; ++c) { float v = xr[c]; ss += v * as[c]; dd += v * ad[c]; }
  s[idx] = ss; d[idx] = dd;
}

__global__ void k_init_attn(unsigned* __restrict__ mu, float* __restrict__ den, int N)
{
  int idx = blockIdx.x * blockDim.x + threadIdx.x;
  if (idx >= N * HEADS) return;
  mu[idx] = fmap(-1e30f);
  den[idx] = 0.f;
}

__global__ void k_init_bias(float* __restrict__ h, const float* __restrict__ b,
                            int N, int On)
{
  int idx = blockIdx.x * blockDim.x + threadIdx.x;
  if (idx >= N * On) return;
  h[idx] = b[idx & (On - 1)];           // On is a power of two (256/128)
}

__device__ __forceinline__ void edge_sd(const int* ei, int E, int e, int& si, int& di)
{
  if (e < E) { si = ei[e]; di = ei[E + e]; }
  else       { si = di = e - E; }       // self-loop
}

__global__ void k_edge_max(const int* __restrict__ ei, int E, int N,
                           const float* __restrict__ s, const float* __restrict__ d,
                           unsigned* __restrict__ mu)
{
  int idx = blockIdx.x * blockDim.x + threadIdx.x;
  int tot = (E + N) * HEADS;
  if (idx >= tot) return;
  int e = idx >> 2, h = idx & 3, si, di;
  edge_sd(ei, E, e, si, di);
  float v = s[si * HEADS + h] + d[di * HEADS + h];
  v = v > 0.f ? v : SLOPE * v;
  atomicMax(&mu[di * HEADS + h], fmap(v));
}

__global__ void k_edge_exp(const int* __restrict__ ei, int E, int N,
                           const float* __restrict__ s, const float* __restrict__ d,
                           const unsigned* __restrict__ mu,
                           float* __restrict__ den, float* __restrict__ ex)
{
  int idx = blockIdx.x * blockDim.x + threadIdx.x;
  int tot = (E + N) * HEADS;
  if (idx >= tot) return;
  int e = idx >> 2, h = idx & 3, si, di;
  edge_sd(ei, E, e, si, di);
  float v = s[si * HEADS + h] + d[di * HEADS + h];
  v = v > 0.f ? v : SLOPE * v;
  float m = funmap(mu[di * HEADS + h]);
  float x = __expf(v - m);
  ex[idx] = x;
  atomicAdd(&den[di * HEADS + h], x);
}

__global__ void k_edge_aggr(const int* __restrict__ ei, int E, int N, int On, int cshift,
                            const float* __restrict__ ex, const float* __restrict__ den,
                            const float* __restrict__ xw, float* __restrict__ outh)
{
  int idx = blockIdx.x * blockDim.x + threadIdx.x;
  int tot = (E + N) * On;
  if (idx >= tot) return;
  int e = idx / On, c = idx - e * On;
  int h = c >> cshift;                  // c / (On/HEADS)
  int si, di;
  edge_sd(ei, E, e, si, di);
  float alpha = ex[e * HEADS + h] / (den[di * HEADS + h] + 1e-16f);
  atomicAdd(&outh[(size_t)di * On + c], alpha * xw[(size_t)si * On + c]);
}

// ---------------------------------------------------------------------------
// Weibull head: k = clip(softplus(kp),.1,1000); l = max(softplus(lp),rm)/Γ(1+1/k)
// theta = clip(mean_s l * (-log(max(1-eps,rm)))^(1/k), rm, 1000)   (all [Z,N])
// ---------------------------------------------------------------------------
__global__ void k_right(const float* __restrict__ kp, const float* __restrict__ lp,
                        const float* __restrict__ eps,
                        float* __restrict__ theta_o, float* __restrict__ k_o,
                        float* __restrict__ l_o, int N, int Z, int S)
{
  int idx = blockIdx.x * blockDim.x + threadIdx.x;
  if (idx >= Z * N) return;
  int zi = idx / N, n = idx - zi * N;
  size_t src = (size_t)n * Z + zi;      // pre-activations are [N,Z]
  float k = softplusf(kp[src]);
  k = fminf(fmaxf(k, KMINF), KMAXF);
  float l = fmaxf(softplusf(lp[src]), REAL_MINF);
  l = l / __expf(lgammaf(1.f + 1.f / k));
  float invk = 1.f / k;
  float acc = 0.f;
  for (int s = 0; s < S; ++s) {
    float e = eps[((size_t)s * Z + zi) * N + n];
    float t = -__logf(fmaxf(1.f - e, REAL_MINF));
    acc += __powf(t, invk);
  }
  float theta = l * acc / (float)S;
  theta = fminf(fmaxf(theta, REAL_MINF), THETA_MAXF);
  size_t dst = (size_t)zi * N + n;
  theta_o[dst] = theta;
  k_o[dst] = k;
  l_o[dst] = l;
}

// ---------------------------------------------------------------------------
static inline void gemm(const bf16_t* A, const bf16_t* W, const float* b, float* out,
                        int M, int K, int On, int act, hipStream_t s)
{
  dim3 blk(128);
  dim3 grd((unsigned)((M / 16 + 3) / 4), (unsigned)(On / (16 * NTILES)));
  if (K == 256)      k_gemm_bf16wmma<256><<<grd, blk, 0, s>>>(A, W, b, out, M, On, act);
  else if (K == 128) k_gemm_bf16wmma<128><<<grd, blk, 0, s>>>(A, W, b, out, M, On, act);
  else               k_gemm_bf16wmma<64> <<<grd, blk, 0, s>>>(A, W, b, out, M, On, act);
}

static inline void f2bf(const float* in, bf16_t* out, int n, hipStream_t s)
{
  int thr = 256;
  int blocks = (n / 8 + thr - 1) / thr;
  k_f2bf<<<blocks, thr, 0, s>>>(in, out, n);
}

extern "C" void kernel_launch(void* const* d_in, const int* in_sizes, int n_in,
                              void* d_out, int out_size, void* d_ws, size_t ws_size,
                              hipStream_t stream)
{
  const float* x     = (const float*)d_in[0];
  const int*   ei    = (const int*)  d_in[1];
  const float* gatW0 = (const float*)d_in[2];
  const float* asrc0 = (const float*)d_in[3];
  const float* adst0 = (const float*)d_in[4];
  const float* gatb0 = (const float*)d_in[5];
  const float* fcW0  = (const float*)d_in[6];
  const float* fcb0  = (const float*)d_in[7];
  const float* shW0  = (const float*)d_in[8];
  const float* shb0  = (const float*)d_in[9];
  const float* scW0  = (const float*)d_in[10];
  const float* scb0  = (const float*)d_in[11];
  const float* gatW1 = (const float*)d_in[12];
  const float* asrc1 = (const float*)d_in[13];
  const float* adst1 = (const float*)d_in[14];
  const float* gatb1 = (const float*)d_in[15];
  const float* fcW1  = (const float*)d_in[16];
  const float* fcb1  = (const float*)d_in[17];
  const float* shW1  = (const float*)d_in[18];
  const float* shb1  = (const float*)d_in[19];
  const float* scW1  = (const float*)d_in[20];
  const float* scb1  = (const float*)d_in[21];
  const float* eps0  = (const float*)d_in[22];
  const float* eps1  = (const float*)d_in[23];

  const int N = in_sizes[0] / 256;
  const int E = in_sizes[1] / 2;
  const int S = in_sizes[22] / (128 * N);
  const int Etot = E + N;

  // ---- workspace layout ----
  float* ws  = (float*)d_ws;
  float* XW  = ws;                               // N*256: xw0 ; later Z0 | XW1
  float* H   = XW  + (size_t)N * 256;            // N*256: h0  ; later H1 | Z1
  float* Sb  = H   + (size_t)N * 256;            // N*4
  float* Db  = Sb  + (size_t)N * HEADS;          // N*4
  unsigned* Mu = (unsigned*)(Db + (size_t)N * HEADS);   // N*4 (uint)
  float* Den = (float*)Mu + (size_t)N * HEADS;   // N*4
  float* EX  = Den + (size_t)N * HEADS;          // Etot*4
  float* KP0 = EX  + (size_t)Etot * HEADS;       // N*128
  float* LP0 = KP0 + (size_t)N * 128;            // N*128
  float* KP1 = LP0 + (size_t)N * 128;            // N*64
  float* LP1 = KP1 + (size_t)N * 64;             // N*64
  // bf16 staging region (16B-aligned)
  size_t foff = (size_t)(LP1 + (size_t)N * 64 - ws);
  foff = (foff + 3) & ~(size_t)3;                // round to 16 bytes
  bf16_t* Xbf  = (bf16_t*)(ws + foff);           // N*256 bf16
  bf16_t* Hbf  = Xbf  + (size_t)N * 256;         // N*256 (reused for h1: N*128)
  bf16_t* Z0bf = Hbf  + (size_t)N * 256;         // N*128
  bf16_t* Z1bf = Z0bf + (size_t)N * 128;         // N*64
  bf16_t* WgatW0 = Z1bf  + (size_t)N * 64;       // 256*256
  bf16_t* WfcW0  = WgatW0 + 256 * 256;           // 128*256
  bf16_t* WgatW1 = WfcW0  + 128 * 256;           // 128*256
  bf16_t* WfcW1  = WgatW1 + 128 * 256;           // 64*128
  bf16_t* WshW0  = WfcW1  + 64 * 128;            // 128*128
  bf16_t* WscW0  = WshW0  + 128 * 128;           // 128*128
  bf16_t* WshW1  = WscW0  + 128 * 128;           // 64*64
  bf16_t* WscW1  = WshW1  + 64 * 64;             // 64*64

  const int THR = 256;
  const int nh  = N * HEADS;
  const int eh  = Etot * HEADS;

  // ---- one-time bf16 conversions (weights + input) ----
  f2bf(x,     Xbf,    N * 256,   stream);
  f2bf(gatW0, WgatW0, 256 * 256, stream);
  f2bf(fcW0,  WfcW0,  128 * 256, stream);
  f2bf(gatW1, WgatW1, 128 * 256, stream);
  f2bf(fcW1,  WfcW1,  64 * 128,  stream);
  f2bf(shW0,  WshW0,  128 * 128, stream);
  f2bf(scW0,  WscW0,  128 * 128, stream);
  f2bf(shW1,  WshW1,  64 * 64,   stream);
  f2bf(scW1,  WscW1,  64 * 64,   stream);

  // ================= layer 0: GAT(256 -> 4x64) =================
  gemm(Xbf, WgatW0, nullptr, XW, N, 256, 256, 0, stream);
  k_srcdst  <<<(nh + THR - 1) / THR, THR, 0, stream>>>(XW, asrc0, adst0, Sb, Db, N, 256);
  k_init_attn<<<(nh + THR - 1) / THR, THR, 0, stream>>>(Mu, Den, N);
  k_init_bias<<<(N * 256 + THR - 1) / THR, THR, 0, stream>>>(H, gatb0, N, 256);
  k_edge_max<<<(eh + THR - 1) / THR, THR, 0, stream>>>(ei, E, N, Sb, Db, Mu);
  k_edge_exp<<<(eh + THR - 1) / THR, THR, 0, stream>>>(ei, E, N, Sb, Db, Mu, Den, EX);
  k_edge_aggr<<<(unsigned)(((size_t)Etot * 256 + THR - 1) / THR), THR, 0, stream>>>(
      ei, E, N, 256, 6, EX, Den, XW, H);

  // z0 = softplus(h0 @ fcW0^T + fcb0); xw1 = h0 @ gatW1^T   (xw0 is dead)
  float* Z0  = XW;
  float* XW1 = XW + (size_t)N * 128;
  f2bf(H, Hbf, N * 256, stream);
  gemm(Hbf, WfcW0,  fcb0,    Z0,  N, 256, 128, 1, stream);
  gemm(Hbf, WgatW1, nullptr, XW1, N, 256, 128, 0, stream);

  // ================= layer 1: GAT(256 -> 4x32) =================
  float* H1 = H;                       // h0 dead from here on
  float* Z1 = H + (size_t)N * 128;
  k_srcdst  <<<(nh + THR - 1) / THR, THR, 0, stream>>>(XW1, asrc1, adst1, Sb, Db, N, 128);
  k_init_attn<<<(nh + THR - 1) / THR, THR, 0, stream>>>(Mu, Den, N);
  k_init_bias<<<(N * 128 + THR - 1) / THR, THR, 0, stream>>>(H1, gatb1, N, 128);
  k_edge_max<<<(eh + THR - 1) / THR, THR, 0, stream>>>(ei, E, N, Sb, Db, Mu);
  k_edge_exp<<<(eh + THR - 1) / THR, THR, 0, stream>>>(ei, E, N, Sb, Db, Mu, Den, EX);
  k_edge_aggr<<<(unsigned)(((size_t)Etot * 128 + THR - 1) / THR), THR, 0, stream>>>(
      ei, E, N, 128, 5, EX, Den, XW1, H1);

  bf16_t* H1bf = Hbf;                  // reuse (h0 bf16 dead now)
  f2bf(H1, H1bf, N * 128, stream);
  gemm(H1bf, WfcW1, fcb1, Z1, N, 128, 64, 1, stream);

  // ================= Weibull heads =================
  f2bf(Z0, Z0bf, N * 128, stream);
  f2bf(Z1, Z1bf, N * 64,  stream);
  gemm(Z0bf, WshW0, shb0, KP0, N, 128, 128, 0, stream);
  gemm(Z0bf, WscW0, scb0, LP0, N, 128, 128, 0, stream);
  gemm(Z1bf, WshW1, shb1, KP1, N, 64, 64, 0, stream);
  gemm(Z1bf, WscW1, scb1, LP1, N, 64, 64, 0, stream);

  // d_out: (theta0, theta1, k0, k1, l0, l1) flat in return order
  float* out    = (float*)d_out;
  float* theta0 = out;
  float* theta1 = out + (size_t)128 * N;
  float* k0     = out + (size_t)192 * N;
  float* k1     = out + (size_t)320 * N;
  float* l0     = out + (size_t)384 * N;
  float* l1     = out + (size_t)512 * N;

  k_right<<<(N * 128 + THR - 1) / THR, THR, 0, stream>>>(KP0, LP0, eps0, theta0, k0, l0, N, 128, S);
  k_right<<<(N * 64  + THR - 1) / THR, THR, 0, stream>>>(KP1, LP1, eps1, theta1, k1, l1, N, 64,  S);
}
